// RNN_3934190043362
// MI455X (gfx1250) — matrix operand order repaired and checked by reading, not compiled, not run on hardware
//
#include <hip/hip_runtime.h>

// ---------------------------------------------------------------------------
// Bidirectional LSTM on gfx1250 (MI455X), bf16 WMMA + f32 accumulate.
//   B=32, T=1024, D=512, H=256, 4H=1024. Output [B,T,2H] f32.
// Phase 0: fp32 -> bf16 conversion of x and weights; bias fold (bih+bhh).
// Phase 1: chip-wide WMMA GEMM  xp = x @ Wih^T + bias (1 M-tile x 4 N-tiles
//          per wave for A reuse).
// Phase 2: 4 persistent workgroups run the sequential scan; recurrent GEMM in
//          WMMA with h/c state in LDS; xp tiles streamed in ahead of time by
//          the Tensor Data Mover (tensor_load_to_lds, double-buffered LDS,
//          s_wait_tensorcnt).
// ---------------------------------------------------------------------------

typedef unsigned short u16;
typedef unsigned int   u32;
typedef unsigned long long u64;
typedef __bf16 v16bf __attribute__((ext_vector_type(16)));
typedef float  v8f   __attribute__((ext_vector_type(8)));
typedef u32    v8u   __attribute__((ext_vector_type(8)));
typedef u32    v4u   __attribute__((ext_vector_type(4)));

#define NB  32
#define NT  1024
#define ND  512
#define NH  256
#define NG  1024   /* 4H */

__device__ __forceinline__ u16 f2bf(float f) {
  u32 u = __float_as_uint(f);
  u32 r = u + 0x7FFFu + ((u >> 16) & 1u);   // round-to-nearest-even
  return (u16)(r >> 16);
}

__device__ __forceinline__ float sigmoidf_(float x) {
  return 1.0f / (1.0f + __expf(-x));
}

// A fragment (16x32 MxK, bf16, row-major source with leading dim ld):
//   lanes 0-15: M=lane, K pairs {0,1},{2,3},{4,5},{6,7},{16,17}..{22,23}
//   lanes 16-31: M=lane-16, same + 8
__device__ __forceinline__ v16bf load_a_frag(const u16* __restrict__ base,
                                             int ld, int m0, int k0, int lane) {
  const int grp = lane >> 4, row = lane & 15;
  v8u w;
#pragma unroll
  for (int v = 0; v < 8; ++v) {
    int kp = (v < 4 ? 2 * v : 16 + 2 * (v - 4)) + 8 * grp;
    w[v] = *(const u32*)(base + (m0 + row) * ld + k0 + kp);
  }
  return __builtin_bit_cast(v16bf, w);
}

// B fragment (32x16 KxN) gathered from row-major W[4H, K] so that
// B[k][n] = W[n0+n, k0+k]:
//   lanes 0-15: N=lane, K 0..7,16..23 ; lanes 16-31: N=lane-16, K 8..15,24..31
__device__ __forceinline__ v16bf load_b_frag(const u16* __restrict__ W,
                                             int ld, int n0, int k0, int lane) {
  const int grp = lane >> 4, col = lane & 15;
  v8u w;
#pragma unroll
  for (int v = 0; v < 8; ++v) {
    int kb = (v < 4 ? 2 * v : 16 + 2 * (v - 4)) + 8 * grp;
    w[v] = *(const u32*)(W + (n0 + col) * ld + k0 + kb);
  }
  return __builtin_bit_cast(v16bf, w);
}

// ---------------------------------------------------------------------------
// TDM: 2D tile load Global -> LDS.  16 rows (stride NT*NG f32) x 1024 f32.
// D# group0/group1 bit layout per CDNA5 ISA 8.3/8.4; groups 2/3 omitted (2D).
// ---------------------------------------------------------------------------
__device__ __forceinline__ void tdm_load_xp_tile(u64 gaddr, u32 lds_off) {
  v4u g0;
  g0[0] = 1u;                                   // count=1 (valid), user bits 0
  g0[1] = lds_off;                              // lds_addr (bytes)
  g0[2] = (u32)gaddr;                           // global_addr[31:0]
  g0[3] = ((u32)(gaddr >> 32) & 0x01FFFFFFu)    // global_addr[56:32]
        | (2u << 30);                           // type=2 ("image")
  v8u g1;
  g1[0] = 2u << 16;                             // data_size=2 -> 4 bytes
  g1[1] = ((u32)NG) << 16;                      // tensor_dim0[15:0]=1024
  g1[2] = 16u << 16;                            // dim0 hi=0 | tensor_dim1=16
  g1[3] = ((u32)NG) << 16;                      // dim1 hi=0 | tile_dim0=1024
  g1[4] = 16u;                                  // tile_dim1=16, tile_dim2=0
  g1[5] = (u32)(NT * NG);                       // tensor_dim0_stride lo32
  g1[6] = 0u;                                   // stride hi | dim1_stride lo
  g1[7] = 0u;                                   // dim1_stride hi
  asm volatile("tensor_load_to_lds %0, %1" :: "s"(g0), "s"(g1) : "memory");
}

// ---------------------------------------------------------------------------
// Phase 0 kernels
// ---------------------------------------------------------------------------
__global__ void k_f2bf(const float* __restrict__ src, u16* __restrict__ dst, int n) {
  int i = blockIdx.x * blockDim.x + threadIdx.x;
  int stride = gridDim.x * blockDim.x;
  for (; i < n; i += stride) dst[i] = f2bf(src[i]);
}

__global__ void k_bias(const float* __restrict__ a, const float* __restrict__ b,
                       const float* __restrict__ c, const float* __restrict__ d,
                       float* __restrict__ bf, float* __restrict__ bb) {
  int i = blockIdx.x * blockDim.x + threadIdx.x;
  if (i < NG) { bf[i] = a[i] + b[i]; bb[i] = c[i] + d[i]; }
}

// ---------------------------------------------------------------------------
// Phase 1: xp[n, g] = sum_d x[n,d] * Wih[g,d] + bias[g], both directions.
// Each wave: 1 M-tile x 4 N-tiles (A fragments reused 4x). K=512 -> 16 steps.
// Task grid per dir: 2048 M-tiles x 16 N-groups = 32768 (2^15) wave tasks.
// ---------------------------------------------------------------------------
__global__ void __launch_bounds__(256) k_xproj(
    const u16* __restrict__ xb,
    const u16* __restrict__ wf, const u16* __restrict__ wb,
    const float* __restrict__ biasf, const float* __restrict__ biasb,
    float* __restrict__ xpf, float* __restrict__ xpb) {
  const int lane = threadIdx.x & 31;
  const int wave = blockIdx.x * (blockDim.x >> 5) + (threadIdx.x >> 5);
  const int dir = wave >> 15;
  const int rem = wave & 32767;
  const int m0 = (rem >> 4) << 4;          // row tile over B*T = 32768
  const int n0g = (rem & 15) * 64;         // 4-tile group over 4H = 1024

  const u16*   W    = dir ? wb : wf;
  const float* bias = dir ? biasb : biasf;
  float*       xp   = dir ? xpb : xpf;

  v8f acc[4] = {};
#pragma unroll
  for (int k0 = 0; k0 < ND; k0 += 32) {
    v16bf a = load_a_frag(xb, ND, m0, k0, lane);
#pragma unroll
    for (int j = 0; j < 4; ++j) {
      v16bf b = load_b_frag(W, ND, n0g + j * 16, k0, lane);
      acc[j] = __builtin_amdgcn_wmma_f32_16x16x32_bf16(false, a, false, b,
                                                       (short)0, acc[j],
                                                       false, false);
    }
  }

  const int grp = lane >> 4, col = lane & 15;
#pragma unroll
  for (int j = 0; j < 4; ++j) {
    const int n0 = n0g + j * 16;
    const float bv = bias[n0 + col];
#pragma unroll
    for (int r = 0; r < 8; ++r) {
      int m = r + grp * 8;                 // C/D layout: vgpr r -> row r (+8)
      xp[(size_t)(m0 + m) * NG + n0 + col] = acc[j][r] + bv;
    }
  }
}

// ---------------------------------------------------------------------------
// Phase 2: persistent scan. blockIdx.x = dir*2 + batch_half.
// 512 threads = 16 waves; each wave owns 4 N-tiles of gates[16, 1024].
// LDS: xp double buffer 2x64KB | gates 64KB | c 16KB | h 8KB  = 216 KB.
// ---------------------------------------------------------------------------
#define XPBUF_BYTES (16 * NG * 4)          /* 65536 */

__global__ void __launch_bounds__(512) k_scan(
    const float* __restrict__ xpf, const float* __restrict__ xpb,
    const u16* __restrict__ whf, const u16* __restrict__ whb,
    float* __restrict__ out) {
  extern __shared__ char smem[];
  float* xpb0  = (float*)(smem);                       // [16][1024]
  float* xpb1  = (float*)(smem + XPBUF_BYTES);         // [16][1024]
  float* gates = (float*)(smem + 2 * XPBUF_BYTES);     // [16][1024]
  float* c_s   = (float*)(smem + 3 * XPBUF_BYTES);     // [16][256]
  u16*   h_s   = (u16*)(smem + 3 * XPBUF_BYTES + 16 * NH * 4); // [16][256]

  const int dir = blockIdx.x >> 1;
  const int bh  = blockIdx.x & 1;
  const u16*   Whh = dir ? whb : whf;
  const float* xp  = dir ? xpb : xpf;
  const u64 xp_base = (u64)(size_t)xp;

  const int tid = threadIdx.x;
  const int lane = tid & 31;
  const int wave = tid >> 5;
  const int grp = lane >> 4, col = lane & 15;

  for (int i = tid; i < 16 * NH; i += 512) { c_s[i] = 0.0f; h_s[i] = 0; }

  // Prologue: TDM-load xp tile for step 0 into buffer 0 (wave 0 issues).
  if (wave == 0) {
    int t0 = dir ? (NT - 1) : 0;
    u64 ga = xp_base + ((u64)(bh * 16) * NT + (u64)t0) * (NG * 4);
    tdm_load_xp_tile(ga, 0u);
  }

  for (int step = 0; step < NT; ++step) {
    const int t = dir ? (NT - 1 - step) : step;
    const int cur = step & 1;
    const float* xp_cur = cur ? xpb1 : xpb0;

    if (wave == 0) {
      // Issue next tile into the other buffer, then ensure current is done.
      int ns = (step + 1 < NT) ? (step + 1) : step;   // clamp (redundant load)
      int tn = dir ? (NT - 1 - ns) : ns;
      u64 ga = xp_base + ((u64)(bh * 16) * NT + (u64)tn) * (NG * 4);
      tdm_load_xp_tile(ga, (u32)((cur ^ 1) * XPBUF_BYTES));
      __builtin_amdgcn_s_wait_tensorcnt(1);
    }
    __syncthreads();   // xp_cur visible to all; h_s/c_s init visible (step 0)

    // gates[16,1024] = h @ Whh^T + xp_cur ; A fragments hoisted (reused 4x)
    v16bf af[8];
#pragma unroll
    for (int i = 0; i < 8; ++i)
      af[i] = load_a_frag(h_s, NH, 0, i * 32, lane);

    for (int q = 0; q < 4; ++q) {
      const int n0 = (wave * 4 + q) * 16;
      v8f acc = {};
#pragma unroll
      for (int i = 0; i < 8; ++i) {
        v16bf b = load_b_frag(Whh, NH, n0, i * 32, lane);
        acc = __builtin_amdgcn_wmma_f32_16x16x32_bf16(false, af[i], false, b,
                                                      (short)0, acc,
                                                      false, false);
      }
#pragma unroll
      for (int r = 0; r < 8; ++r) {
        int m = r + grp * 8;
        gates[m * NG + n0 + col] = acc[r] + xp_cur[m * NG + n0 + col];
      }
    }
    __syncthreads();

    // elementwise gate math + state update; unit u = b_local*256 + j
#pragma unroll
    for (int q = 0; q < 8; ++q) {
      int u = tid + q * 512;               // 0..4095
      int bl = u >> 8, jj = u & 255;
      float iv = sigmoidf_(gates[bl * NG + jj]);
      float fv = sigmoidf_(gates[bl * NG + NH + jj]);
      float gv = tanhf    (gates[bl * NG + 2 * NH + jj]);
      float ov = sigmoidf_(gates[bl * NG + 3 * NH + jj]);
      float c = fv * c_s[u] + iv * gv;
      c_s[u] = c;
      float h = ov * tanhf(c);
      h_s[u] = f2bf(h);
      out[(size_t)((bh * 16 + bl) * NT + t) * (2 * NH) + dir * NH + jj] = h;
    }
    __syncthreads();
  }
}

// ---------------------------------------------------------------------------
extern "C" void kernel_launch(void* const* d_in, const int* in_sizes, int n_in,
                              void* d_out, int out_size, void* d_ws, size_t ws_size,
                              hipStream_t stream) {
  const float* x     = (const float*)d_in[0];
  const float* Wih_f = (const float*)d_in[1];
  const float* Whh_f = (const float*)d_in[2];
  const float* bih_f = (const float*)d_in[3];
  const float* bhh_f = (const float*)d_in[4];
  const float* Wih_b = (const float*)d_in[5];
  const float* Whh_b = (const float*)d_in[6];
  const float* bih_b = (const float*)d_in[7];   // "bib_b" in reference
  const float* bhh_b = (const float*)d_in[8];

  char* ws = (char*)d_ws;
  size_t off = 0;
  u16* xbf   = (u16*)(ws + off); off += (size_t)NB * NT * ND * 2;   // 33.5 MB
  u16* wihf  = (u16*)(ws + off); off += (size_t)NG * ND * 2;        // 1 MB
  u16* wihb  = (u16*)(ws + off); off += (size_t)NG * ND * 2;
  u16* whhf  = (u16*)(ws + off); off += (size_t)NG * NH * 2;        // 0.5 MB
  u16* whhb  = (u16*)(ws + off); off += (size_t)NG * NH * 2;
  float* bsf = (float*)(ws + off); off += (size_t)NG * 4;
  float* bsb = (float*)(ws + off); off += (size_t)NG * 4;
  float* xpf = (float*)(ws + off); off += (size_t)NB * NT * NG * 4; // 134 MB
  float* xpb = (float*)(ws + off); off += (size_t)NB * NT * NG * 4; // 134 MB

  // Phase 0: conversions (stream order covers dependencies)
  k_f2bf<<<8192, 256, 0, stream>>>(x, xbf, NB * NT * ND);
  k_f2bf<<<512, 256, 0, stream>>>(Wih_f, wihf, NG * ND);
  k_f2bf<<<512, 256, 0, stream>>>(Wih_b, wihb, NG * ND);
  k_f2bf<<<256, 256, 0, stream>>>(Whh_f, whhf, NG * NH);
  k_f2bf<<<256, 256, 0, stream>>>(Whh_b, whhb, NG * NH);
  k_bias<<<NG / 256, 256, 0, stream>>>(bih_f, bhh_f, bih_b, bhh_b, bsf, bsb);

  // Phase 1: input projection GEMM, both directions (65536 wave tasks, 8/blk)
  k_xproj<<<8192, 256, 0, stream>>>(xbf, wihf, wihb, bsf, bsb, xpf, xpb);

  // Phase 2: recurrent scan, 4 persistent workgroups, 216 KB LDS each
  size_t shm = (size_t)3 * XPBUF_BYTES + (size_t)16 * NH * 4 + (size_t)16 * NH * 2;
  k_scan<<<4, 512, shm, stream>>>(xpf, xpb, whhf, whhb, (float*)d_out);
}
